// SelfAttention_29549374996784
// MI455X (gfx1250) — matrix operand order repaired and checked by reading, not compiled, not run on hardware
//
#include <hip/hip_runtime.h>

#define BATCH 8
#define CIN   256
#define NN    1024
#define OCH   384          // 256 q-ch + 64 k-ch + 64 v-ch, fused projection
#define SFACT 0.35355339059327373f   // 64^-0.25
#define EPSV  1e-5f
#define RPE_ELEMS (63 * 63 * 64)     // 254016

typedef __attribute__((ext_vector_type(16))) __bf16 v16bf;
typedef __attribute__((ext_vector_type(8)))  float  v8f;

union ABPack { uint4 u[2]; v16bf v; };

__device__ __forceinline__ unsigned short f2bf(float f) {
  unsigned int u = __float_as_uint(f);
  unsigned int r = u + 0x7FFFu + ((u >> 16) & 1u);   // round-to-nearest-even
  return (unsigned short)(r >> 16);
}

// ---------------------------------------------------------------------------
// K0: pre-convert the rpe table to bf16 once (halves gather traffic in k_pos)
// ---------------------------------------------------------------------------
__global__ void k_rpe_pack(const float* __restrict__ rpe,
                           unsigned short* __restrict__ rpe_bf) {
  int i = blockIdx.x * 256 + threadIdx.x;
  if (i < RPE_ELEMS) rpe_bf[i] = f2bf(rpe[i]);
}

// ---------------------------------------------------------------------------
// K1: fused projection GEMM  P[b,o,n] = sum_c W[o,c] * x[b,c,n]
//     o: 0..255 -> Wq, 256..319 -> Wk, 320..383 -> Wv
// ---------------------------------------------------------------------------
__global__ void k_proj(const float* __restrict__ x,
                       const float* __restrict__ Wq,
                       const float* __restrict__ Wk,
                       const float* __restrict__ Wv,
                       float* __restrict__ P) {
  const int n0 = blockIdx.x * 64;
  const int o0 = blockIdx.y * 16;
  const int b  = blockIdx.z;
  const int t  = threadIdx.x;
  __shared__ float xs[64][64];
  __shared__ float wt[16][64];
  const int o_l = t >> 4;           // 0..15
  const int n_l = (t & 15) * 4;     // 0..60
  float acc[4] = {0.f, 0.f, 0.f, 0.f};
  for (int c0 = 0; c0 < CIN; c0 += 64) {
    __syncthreads();
    for (int idx = t; idx < 4096; idx += 256) {
      int cc = idx >> 6, nn = idx & 63;
      xs[cc][nn] = x[(size_t)(b * CIN + c0 + cc) * NN + n0 + nn];
    }
    for (int idx = t; idx < 1024; idx += 256) {
      int oo = idx >> 6, cc = idx & 63;
      int o = o0 + oo;
      const float* row = (o < 256) ? (Wq + (size_t)o * 256)
                       : (o < 320) ? (Wk + (size_t)(o - 256) * 256)
                                   : (Wv + (size_t)(o - 320) * 256);
      wt[oo][cc] = row[c0 + cc];
    }
    __syncthreads();
    for (int cc = 0; cc < 64; ++cc) {
      float wv = wt[o_l][cc];
      acc[0] += wv * xs[cc][n_l + 0];
      acc[1] += wv * xs[cc][n_l + 1];
      acc[2] += wv * xs[cc][n_l + 2];
      acc[3] += wv * xs[cc][n_l + 3];
    }
  }
  float* dst = P + (size_t)(b * OCH + o0 + o_l) * NN + n0 + n_l;
  dst[0] = acc[0]; dst[1] = acc[1]; dst[2] = acc[2]; dst[3] = acc[3];
}

// ---------------------------------------------------------------------------
// K2: BatchNorm stats per channel (0..319): mean + rsqrt(var+eps) over (b,n)
// ---------------------------------------------------------------------------
__global__ void k_stats(const float* __restrict__ P, float* __restrict__ stats) {
  const int c = blockIdx.x;     // 0..319
  const int t = threadIdx.x;
  float s = 0.f, s2 = 0.f;
  for (int idx = t; idx < 8192; idx += 256) {
    int b = idx >> 10, n = idx & 1023;
    float v = P[(size_t)(b * OCH + c) * NN + n];
    s += v; s2 += v * v;
  }
  __shared__ float rs[256], rs2[256];
  rs[t] = s; rs2[t] = s2; __syncthreads();
  for (int off = 128; off; off >>= 1) {
    if (t < off) { rs[t] += rs[t + off]; rs2[t] += rs2[t + off]; }
    __syncthreads();
  }
  if (t == 0) {
    float mu  = rs[0] * (1.f / 8192.f);
    float var = rs2[0] * (1.f / 8192.f) - mu * mu;
    stats[c]       = mu;
    stats[OCH + c] = rsqrtf(var + EPSV);
  }
}

// ---------------------------------------------------------------------------
// K3: (a) normalize q channels in place (BN * gamma + beta, then *SF)
//     (b) pack v channels to bf16, v-major layout vbf[b][v][m]
// ---------------------------------------------------------------------------
__global__ void k_post(float* __restrict__ P, const float* __restrict__ stats,
                       const float* __restrict__ gq, const float* __restrict__ bq,
                       unsigned short* __restrict__ vbf) {
  size_t gid = (size_t)blockIdx.x * 256 + threadIdx.x;
  const size_t NQ = (size_t)BATCH * 256 * NN;    // 2097152
  if (gid < NQ) {
    int b = (int)(gid >> 18);
    int c = (int)(gid >> 10) & 255;
    int n = (int)gid & 1023;
    size_t idx = (size_t)(b * OCH + c) * NN + n;
    float v = P[idx];
    P[idx] = ((v - stats[c]) * stats[OCH + c] * gq[c] + bq[c]) * SFACT;
  } else {
    size_t g2 = gid - NQ;                        // < 524288
    int b   = (int)(g2 >> 16);
    int vch = (int)(g2 >> 10) & 63;
    int m   = (int)g2 & 1023;
    float v = P[(size_t)(b * OCH + 320 + vch) * NN + m];
    vbf[(size_t)(b * 64 + vch) * NN + m] = f2bf(v);
  }
}

// ---------------------------------------------------------------------------
// K4: softmax over n of BN-normalized, SF-scaled k channels -> ks[b][kc][n]
// ---------------------------------------------------------------------------
__global__ void k_softmax(const float* __restrict__ P, const float* __restrict__ stats,
                          const float* __restrict__ gk, const float* __restrict__ bk,
                          float* __restrict__ ks) {
  const int b  = blockIdx.x >> 6;
  const int kc = blockIdx.x & 63;
  const int c  = 256 + kc;
  const int t  = threadIdx.x;
  const float mu = stats[c], rstd = stats[OCH + c];
  const float g = gk[kc], be = bk[kc];
  const float* row = P + (size_t)(b * OCH + c) * NN;
  float vals[4]; float mx = -1e30f;
  #pragma unroll
  for (int j = 0; j < 4; ++j) {
    float v = ((row[t * 4 + j] - mu) * rstd * g + be) * SFACT;
    vals[j] = v; mx = fmaxf(mx, v);
  }
  __shared__ float red[256];
  red[t] = mx; __syncthreads();
  for (int off = 128; off; off >>= 1) {
    if (t < off) red[t] = fmaxf(red[t], red[t + off]);
    __syncthreads();
  }
  mx = red[0]; __syncthreads();
  float s = 0.f;
  #pragma unroll
  for (int j = 0; j < 4; ++j) { vals[j] = __expf(vals[j] - mx); s += vals[j]; }
  red[t] = s; __syncthreads();
  for (int off = 128; off; off >>= 1) {
    if (t < off) red[t] += red[t + off];
    __syncthreads();
  }
  float inv = 1.f / red[0];
  float* o = ks + (size_t)(b * 64 + kc) * NN;
  #pragma unroll
  for (int j = 0; j < 4; ++j) o[t * 4 + j] = vals[j] * inv;
}

// ---------------------------------------------------------------------------
// K5: content lambda  ctx[b,k,v] = sum_n ks[b,k,n] * v[b,v,n]   (67 MFLOP)
// ---------------------------------------------------------------------------
__global__ void k_context(const float* __restrict__ ks, const float* __restrict__ P,
                          float* __restrict__ ctx) {
  const int b   = blockIdx.y;
  const int pid = blockIdx.x * 256 + threadIdx.x;   // 0..4095 -> (k,v)
  const int kk = pid >> 6, vv = pid & 63;
  const float* kr = ks + (size_t)(b * 64 + kk) * NN;
  const float* vr = P  + (size_t)(b * OCH + 320 + vv) * NN;
  float s = 0.f;
  for (int n = 0; n < NN; ++n) s += kr[n] * vr[n];
  ctx[(size_t)b * 4096 + pid] = s;
}

// ---------------------------------------------------------------------------
// K6: positional lambda + fused epilogue (the 68.7 GFLOP workhorse).
// One block per output position n. 8 waves; wave b accumulates
//   PC_b[k,v] = sum_m rg[n,m,k] * v[b,v,m]        (4x4 WMMA tiles, K=32 steps)
// with rg[n] gathered (bf16) from the pre-packed rpe table -> LDS transpose.
// Epilogue: out[b, h*64+v, n] = sum_k q[b,h,k,n] * (ctx[b,k,v] + PC_b[k,v])
// ---------------------------------------------------------------------------
__global__ void __launch_bounds__(256) k_pos(
    const unsigned short* __restrict__ rpe_bf,
    const unsigned short* __restrict__ vbf,
    const float* __restrict__ P,        // channels 0..255 hold normalized q
    const float* __restrict__ ctx,
    float* __restrict__ out) {
  const int n    = blockIdx.x;
  const int i1   = n >> 5, j1 = n & 31;
  const int t    = threadIdx.x;
  const int b    = t >> 5;        // wave index == batch
  const int lane = t & 31;
  const int lrow = lane & 15;
  const int half = lane >> 4;

  __shared__ unsigned short lds_rg[64 * 32];   // [k][m_local], 4 KB

  v8f acc[4][4];
  const v8f vzero = {0.f, 0.f, 0.f, 0.f, 0.f, 0.f, 0.f, 0.f};
  #pragma unroll
  for (int i = 0; i < 4; ++i)
    #pragma unroll
    for (int j = 0; j < 4; ++j) acc[i][j] = vzero;

  for (int m0 = 0; m0 < NN; m0 += 32) {
    __syncthreads();
    {   // gather rg tile: 64 k x 32 m, transposed into LDS (already bf16)
      const int m_l = t & 31;
      const int kb  = (t >> 5) * 8;
      const int m  = m0 + m_l;
      const int i2 = m >> 5, j2 = m & 31;
      const int ri = i2 - i1 + 31, rj = j2 - j1 + 31;   // rel = grid[m]-grid[n]+31
      const uint4 g = *(const uint4*)(rpe_bf + (size_t)(ri * 63 + rj) * 64 + kb);
      lds_rg[(kb + 0) * 32 + m_l] = (unsigned short)(g.x & 0xFFFFu);
      lds_rg[(kb + 1) * 32 + m_l] = (unsigned short)(g.x >> 16);
      lds_rg[(kb + 2) * 32 + m_l] = (unsigned short)(g.y & 0xFFFFu);
      lds_rg[(kb + 3) * 32 + m_l] = (unsigned short)(g.y >> 16);
      lds_rg[(kb + 4) * 32 + m_l] = (unsigned short)(g.z & 0xFFFFu);
      lds_rg[(kb + 5) * 32 + m_l] = (unsigned short)(g.z >> 16);
      lds_rg[(kb + 6) * 32 + m_l] = (unsigned short)(g.w & 0xFFFFu);
      lds_rg[(kb + 7) * 32 + m_l] = (unsigned short)(g.w >> 16);
    }
    __syncthreads();

    // A tiles (rg^T): lane = M row (k), halves hold m 0-15 / 16-31 contiguous
    ABPack A[4];
    #pragma unroll
    for (int tk = 0; tk < 4; ++tk) {
      const uint4* bp = (const uint4*)&lds_rg[(tk * 16 + lrow) * 32];
      A[tk].u[0] = bp[2 * half];
      A[tk].u[1] = bp[2 * half + 1];
    }
    // B tiles (V_b^T): lane = N col (v), halves hold m 0-15 / 16-31 contiguous
    #pragma unroll
    for (int tv = 0; tv < 4; ++tv) {
      ABPack Bt;
      const uint4* gp = (const uint4*)(vbf +
          (size_t)(b * 64 + tv * 16 + lrow) * NN + m0 + half * 16);
      Bt.u[0] = gp[0];
      Bt.u[1] = gp[1];
      #pragma unroll
      for (int tk = 0; tk < 4; ++tk) {
        acc[tk][tv] = __builtin_amdgcn_wmma_f32_16x16x32_bf16(
            false, A[tk].v, false, Bt.v, (short)0, acc[tk][tv], false, false);
      }
    }
  }

  // fold in content lambda: acc += ctx[b,k,v]
  #pragma unroll
  for (int tk = 0; tk < 4; ++tk) {
    #pragma unroll
    for (int r = 0; r < 8; ++r) {
      const int k = tk * 16 + r + 8 * half;
      #pragma unroll
      for (int tv = 0; tv < 4; ++tv)
        acc[tk][tv][r] += ctx[(size_t)b * 4096 + k * 64 + tv * 16 + lrow];
    }
  }

  // epilogue: out[b, h*64+v, n] = sum_k q[b,h,k,n] * acc[k,v]
  #pragma unroll
  for (int h = 0; h < 4; ++h) {
    float part[4] = {0.f, 0.f, 0.f, 0.f};
    #pragma unroll
    for (int tk = 0; tk < 4; ++tk) {
      #pragma unroll
      for (int r = 0; r < 8; ++r) {
        const int k = tk * 16 + r + 8 * half;
        const float qv = P[(size_t)(b * OCH + h * 64 + k) * NN + n];
        #pragma unroll
        for (int tv = 0; tv < 4; ++tv) part[tv] += qv * acc[tk][tv][r];
      }
    }
    #pragma unroll
    for (int tv = 0; tv < 4; ++tv) {
      float tot = part[tv] + __shfl_xor(part[tv], 16, 32);
      if (half == 0)
        out[(size_t)(b * 256 + h * 64 + tv * 16 + lrow) * NN + n] = tot;
    }
  }
}

// ---------------------------------------------------------------------------
extern "C" void kernel_launch(void* const* d_in, const int* in_sizes, int n_in,
                              void* d_out, int out_size, void* d_ws, size_t ws_size,
                              hipStream_t stream) {
  const float* x   = (const float*)d_in[0];
  const float* Wq  = (const float*)d_in[1];
  const float* gq  = (const float*)d_in[2];
  const float* bq  = (const float*)d_in[3];
  const float* Wk  = (const float*)d_in[4];
  const float* gk  = (const float*)d_in[5];
  const float* bk  = (const float*)d_in[6];
  const float* Wv  = (const float*)d_in[7];
  const float* rpe = (const float*)d_in[8];
  float* out = (float*)d_out;

  char* ws = (char*)d_ws;
  float*          P     = (float*)(ws);                     // 8*384*1024 f32 = 12 MB
  float*          ks    = (float*)(ws + 12582912);          // 8*64*1024 f32  = 2 MB
  unsigned short* vbf   = (unsigned short*)(ws + 14680064); // 8*64*1024 bf16 = 1 MB
  float*          ctx   = (float*)(ws + 15728640);          // 8*64*64 f32
  float*          stat  = (float*)(ws + 15859712);          // 2*384 f32
  unsigned short* rpebf = (unsigned short*)(ws + 15862784); // 63*63*64 bf16 ~ 0.5 MB

  k_rpe_pack<<<dim3((RPE_ELEMS + 255) / 256), 256, 0, stream>>>(rpe, rpebf);
  k_proj    <<<dim3(16, 24, 8), 256, 0, stream>>>(x, Wq, Wk, Wv, P);
  k_stats   <<<dim3(320),       256, 0, stream>>>(P, stat);
  k_post    <<<dim3(10240),     256, 0, stream>>>(P, stat, gq, bq, vbf);
  k_softmax <<<dim3(512),       256, 0, stream>>>(P, stat, gk, bk, ks);
  k_context <<<dim3(16, 8),     256, 0, stream>>>(ks, P, ctx);
  k_pos     <<<dim3(1024),      256, 0, stream>>>(rpebf, vbf, P, ctx, out);
}
